// ScaledDotProductAttention_25993142075924
// MI455X (gfx1250) — compile-verified, matrix operand order checked by reading, not run on hardware
//
#include <hip/hip_runtime.h>

typedef __attribute__((ext_vector_type(16))) _Float16 v16h;
typedef __attribute__((ext_vector_type(8)))  _Float16 v8h;
typedef __attribute__((ext_vector_type(4)))  _Float16 v4h;
typedef __attribute__((ext_vector_type(8)))  float    v8f;

#define B_ 2
#define H_ 16
#define S_ 2048
#define D_ 64
// (1/sqrt(64)) * log2(e): softmax scale folded into exp2 units, applied to Q
#define EXP2_SCALE_ 0.18033688011112042f

// Build a 16-half WMMA fragment from two contiguous 8-half (16B) LDS groups.
// CDNA5 16-bit A/B layout: lane half hf holds K = {0..7}+8hf and {16..23}+8hf
// (plus 32 per K-chunk), so each group is one aligned ds_load_b128.
__device__ __forceinline__ v16h ldsFragB(const _Float16* p_lo, const _Float16* p_hi) {
    v8h lo = *(const v8h*)p_lo;
    v8h hi = *(const v8h*)p_hi;
    return __builtin_shufflevector(lo, hi, 0, 1, 2, 3, 4, 5, 6, 7,
                                           8, 9, 10, 11, 12, 13, 14, 15);
}

__global__ __launch_bounds__(256)
void fa_causal_kernel(const float* __restrict__ Qg, const float* __restrict__ Kg,
                      const float* __restrict__ Vg, float* __restrict__ Og) {
    // K tile: [key][d] f16, stride 72 halves (144 B: 16B-aligned rows, conflict-free)
    // V tile: [d][key] f16 (transposed), stride 72
    // P relayout: per-wave 16x32 f16, stride 40 halves (80 B)
    __shared__ __align__(16) _Float16 Ks[64 * 72];
    __shared__ __align__(16) _Float16 Vs[64 * 72];
    __shared__ __align__(16) _Float16 Ps[8][16 * 40];

    const int qt   = blockIdx.x;       // 128-row query tile
    const int bh   = blockIdx.y;       // b*H + h
    const int tid  = threadIdx.x;
    const int wave = tid >> 5;
    const int lane = tid & 31;
    const int hf   = lane >> 4;        // 16-lane half of the wave
    const int l16  = lane & 15;

    const size_t base = (size_t)bh * (S_ * D_);
    const float* Q = Qg + base;
    const float* K = Kg + base;
    const float* V = Vg + base;
    float*       O = Og + base;

    const int q0 = qt * 128 + wave * 16;   // this wave's first query row

    // ---- Q strip -> two A fragments, pre-scaled by (1/sqrt(D))*log2(e) so the
    // score WMMAs produce exp2-ready values directly (no per-score multiply).
    v16h aQ[2];
    {
        const float* qp = Q + (size_t)(q0 + l16) * D_;
        #pragma unroll
        for (int c = 0; c < 2; ++c) {
            #pragma unroll
            for (int j = 0; j < 8; ++j) {
                aQ[c][j]     = (_Float16)(qp[8 * hf + 32 * c + j] * EXP2_SCALE_);
                aQ[c][8 + j] = (_Float16)(qp[16 + 8 * hf + 32 * c + j] * EXP2_SCALE_);
            }
        }
    }

    // All-ones B fragment: lacc = P . 1 computes row sums on the matrix pipe,
    // replicated across all 16 columns (= all lanes of a half) -- exactly the
    // replication layout the epilogue needs.
    v16h bOnes;
    #pragma unroll
    for (int j = 0; j < 16; ++j) bOnes[j] = (_Float16)1.0f;

    v8f acc[4];   // O tile: 16 rows x 64 d
    v8f lacc;     // softmax denominators (replicated)
    #pragma unroll
    for (int r = 0; r < 8; ++r) {
        lacc[r] = 0.0f;
        #pragma unroll
        for (int t = 0; t < 4; ++t) acc[t][r] = 0.0f;
    }

    // ---- Software-pipelined K/V tile staging (64 keys x 64 d per tile).
    float4 krg[4], vrg[4];
    auto tileLoad = [&](int kb) {
        #pragma unroll
        for (int it = 0; it < 4; ++it) {
            int i   = tid + (it << 8);
            int row = i >> 4;
            int d4  = (i & 15) << 2;
            krg[it] = *(const float4*)(K + (size_t)(kb + row) * D_ + d4);
            vrg[it] = *(const float4*)(V + (size_t)(kb + row) * D_ + d4);
        }
    };
    auto tileStore = [&]() {
        #pragma unroll
        for (int it = 0; it < 4; ++it) {
            int i   = tid + (it << 8);
            int row = i >> 4;
            int d4  = (i & 15) << 2;
            v4h kh;
            kh[0] = (_Float16)krg[it].x; kh[1] = (_Float16)krg[it].y;
            kh[2] = (_Float16)krg[it].z; kh[3] = (_Float16)krg[it].w;
            *(v4h*)(Ks + row * 72 + d4) = kh;                // one ds_store_b64
            Vs[(d4 + 0) * 72 + row] = (_Float16)vrg[it].x;   // transposed scatter
            Vs[(d4 + 1) * 72 + row] = (_Float16)vrg[it].y;
            Vs[(d4 + 2) * 72 + row] = (_Float16)vrg[it].z;
            Vs[(d4 + 3) * 72 + row] = (_Float16)vrg[it].w;
        }
    };

    const int nkt = 2 * qt + 2;   // causal: only tiles with kb <= qt*128+127
    tileLoad(0);

    for (int kt = 0; kt < nkt; ++kt) {
        const int kb = kt * 64;
        tileStore();
        __syncthreads();
        if (kt + 1 < nkt) tileLoad(kb + 64);   // prefetch next tile during compute

        if (kb <= q0 + 15) {                   // wave-uniform causal tile skip
            #pragma unroll
            for (int hh = 0; hh < 2; ++hh) {   // 32-key halves
                if (kb + 32 * hh > q0 + 15) continue;   // uniform sub-skip on diagonal

                // ---- scores S = (scaled Q) . K^T  (two 16-key subtiles)
                v8f s[2];
                #pragma unroll
                for (int js = 0; js < 2; ++js) {
                    const _Float16* kr = Ks + (32 * hh + 16 * js + l16) * 72;
                    v8f cz;
                    #pragma unroll
                    for (int r = 0; r < 8; ++r) cz[r] = 0.0f;
                    #pragma unroll
                    for (int c = 0; c < 2; ++c) {
                        v16h bK = ldsFragB(kr + 8 * hf + 32 * c,
                                           kr + 16 + 8 * hf + 32 * c);
                        cz = __builtin_amdgcn_wmma_f32_16x16x32_f16(
                                 false, aQ[c], false, bK, (short)0, cz, false, false);
                    }
                    s[js] = cz;
                }

                // ---- p = exp2(s) with causal mask; raw v_exp_f32 (flush-to-zero
                // tail is exactly right: masked -inf -> 0). Written straight to
                // the per-wave LDS relayout buffer as f16 (C layout -> row-major).
                const bool mask2 = (kb + 32 * hh + 31) > q0;
                _Float16* pb = Ps[wave];
                #pragma unroll
                for (int js = 0; js < 2; ++js) {
                    #pragma unroll
                    for (int r = 0; r < 8; ++r) {
                        float sv = s[js][r];
                        if (mask2) {
                            int qr = q0 + r + 8 * hf;
                            int kc = kb + 32 * hh + 16 * js + l16;
                            if (kc > qr) sv = -__builtin_inff();
                        }
                        pb[(r + 8 * hf) * 40 + 16 * js + l16] =
                            (_Float16)__builtin_amdgcn_exp2f(sv);
                    }
                }
                asm volatile("" ::: "memory");  // LDS is in-order per wave; fence the compiler
                const _Float16* pr = pb + l16 * 40;
                v16h aP = ldsFragB(pr + 8 * hf, pr + 16 + 8 * hf);

                // ---- O += P . V ; denominators += P . 1 (all on the WMMA pipe)
                #pragma unroll
                for (int t = 0; t < 4; ++t) {
                    const _Float16* vr = Vs + (t * 16 + l16) * 72 + 32 * hh;
                    v16h bV = ldsFragB(vr + 8 * hf, vr + 16 + 8 * hf);
                    acc[t] = __builtin_amdgcn_wmma_f32_16x16x32_f16(
                                 false, aP, false, bV, (short)0, acc[t], false, false);
                }
                lacc = __builtin_amdgcn_wmma_f32_16x16x32_f16(
                           false, aP, false, bOnes, (short)0, lacc, false, false);
            }
        }
        __syncthreads();
    }

    // ---- epilogue: normalize (v_rcp_f32) and store (row = r+8*hf, col = t*16+l16)
    #pragma unroll
    for (int r = 0; r < 8; ++r) {
        float rl = __builtin_amdgcn_rcpf(lacc[r]);
        int qr = q0 + r + 8 * hf;
        float* op = O + (size_t)qr * D_ + l16;
        #pragma unroll
        for (int t = 0; t < 4; ++t) op[t * 16] = acc[t][r] * rl;
    }
}

extern "C" void kernel_launch(void* const* d_in, const int* in_sizes, int n_in,
                              void* d_out, int out_size, void* d_ws, size_t ws_size,
                              hipStream_t stream) {
    (void)in_sizes; (void)n_in; (void)out_size; (void)d_ws; (void)ws_size;
    const float* q = (const float*)d_in[0];
    const float* k = (const float*)d_in[1];
    const float* v = (const float*)d_in[2];
    // d_in[3] is the causal mask; causality is computed analytically in-kernel.
    float* out = (float*)d_out;
    dim3 grid(S_ / 128, B_ * H_);
    fa_causal_kernel<<<grid, dim3(256), 0, stream>>>(q, k, v, out);
}
